// MultiheadAttention_73117523247104
// MI455X (gfx1250) — compile-verified, matrix operand order checked by reading, not run on hardware
//
#include <hip/hip_runtime.h>

// ---------------------------------------------------------------------------
// MI455X (gfx1250) multi-head attention forward.
// bf16 WMMA for all matmuls; async global->LDS staging (ASYNCcnt) with
// double buffering in the GEMMs.
// ---------------------------------------------------------------------------

typedef __attribute__((ext_vector_type(16))) __bf16          bf16x16;
typedef __attribute__((ext_vector_type(8)))  __bf16          bf16x8;
typedef __attribute__((ext_vector_type(8)))  unsigned short  u16x8;
typedef __attribute__((ext_vector_type(8)))  float           f32x8;

static constexpr int S_  = 1025;
static constexpr int B_  = 8;
static constexpr int E_  = 1024;
static constexpr int H_  = 16;
static constexpr int HD_ = 64;

// fp32 -> bf16 (round to nearest even)
__device__ __forceinline__ unsigned short f2bf(float f) {
    unsigned int u = __float_as_uint(f);
    unsigned int r = u + 0x7fffu + ((u >> 16) & 1u);
    return (unsigned short)(r >> 16);
}

// CDNA5 async copy: 16B global -> LDS, tracked by ASYNCcnt.
// lds_addr is the 32-bit LDS byte address (low bits of the generic pointer).
__device__ __forceinline__ void async_ld_b128(unsigned lds_addr, const void* g) {
    asm volatile("global_load_async_to_lds_b128 %0, %1, off"
                 :: "v"(lds_addr), "v"((unsigned long long)g)
                 : "memory");
}
__device__ __forceinline__ void wait_async0() {
    asm volatile("s_wait_asynccnt 0x0" ::: "memory");
}
__device__ __forceinline__ unsigned lds_addr_of(const void* p) {
    return (unsigned)(unsigned long long)p;
}

// Build a 16-element bf16 A/B fragment from two 16-byte LDS chunks.
__device__ __forceinline__ bf16x16 load_frag(const unsigned short* base, int c0, int c1) {
    bf16x8 lo = *(const bf16x8*)(base + c0);
    bf16x8 hi = *(const bf16x8*)(base + c1);
    return __builtin_shufflevector(lo, hi, 0,1,2,3,4,5,6,7,8,9,10,11,12,13,14,15);
}

// ---------------------------------------------------------------------------
// Kernel 1: elementwise fp32 -> bf16
// ---------------------------------------------------------------------------
__global__ void convert_f32_bf16(const float* __restrict__ in,
                                 unsigned short* __restrict__ out,
                                 long long n) {
    long long i      = (long long)blockIdx.x * blockDim.x + threadIdx.x;
    long long stride = (long long)gridDim.x * blockDim.x;
    for (; i < n; i += stride) out[i] = f2bf(in[i]);
}

// ---------------------------------------------------------------------------
// Kernel 2/5: tiled bf16 GEMM  C[M,N] = A[M,K] @ W[N,K]^T + bias
//   mode 0: scatter bf16 into [B,H,S,hd] layout (QKV projection, N==1024)
//   mode 1: fp32 store to fout[M,N]            (output projection)
// Block tile 128x128, 8 waves of 32x64, K step 32, double-buffered async LDS.
// ---------------------------------------------------------------------------
__global__ __launch_bounds__(256)
void gemm_bf16(const unsigned short* __restrict__ A,
               const unsigned short* __restrict__ W,
               const float* __restrict__ bias,
               int M, int N, int K,
               unsigned short* __restrict__ bf16_dst,  // mode 0
               float* __restrict__ fout,               // mode 1
               int mode) {
    __shared__ __attribute__((aligned(16))) unsigned short Al[2][128 * 40];
    __shared__ __attribute__((aligned(16))) unsigned short Bl[2][128 * 40];

    const int tid  = threadIdx.x;
    const int lane = tid & 31;
    const int wave = tid >> 5;
    const int wm   = wave >> 1;   // 0..3 : 32 rows each
    const int wn   = wave & 1;    // 0..1 : 64 cols each
    const int bm   = blockIdx.x;
    const int bn   = blockIdx.y;
    const int r    = lane & 15;
    const int half = lane >> 4;
    const int c0   = half * 8;
    const int c1   = half * 8 + 16;

    f32x8 zero8 = {0.f,0.f,0.f,0.f,0.f,0.f,0.f,0.f};
    f32x8 acc[2][4];
    #pragma unroll
    for (int i = 0; i < 2; i++)
        #pragma unroll
        for (int j = 0; j < 4; j++) acc[i][j] = zero8;

    // per-thread staging coordinates: 4 x 16B chunks (2 for A, 2 for W)
    const int srow = tid >> 2;          // 0..63
    const int scol = (tid & 3) * 8;     // 0,8,16,24

    auto stage = [&](int buf, int k0) {
        unsigned aB = lds_addr_of(&Al[buf][0]);
        unsigned bB = lds_addr_of(&Bl[buf][0]);
        #pragma unroll
        for (int i = 0; i < 2; i++) {
            int row = srow + i * 64;
            long long grow = (long long)bm * 128 + row;
            if (grow >= M) grow = M - 1;
            async_ld_b128(aB + (unsigned)(row * 40 + scol) * 2,
                          &A[grow * K + k0 + scol]);
            long long gn = (long long)bn * 128 + row;
            async_ld_b128(bB + (unsigned)(row * 40 + scol) * 2,
                          &W[gn * K + k0 + scol]);
        }
    };

    const int nk = K / 32;
    stage(0, 0);
    for (int kt = 0; kt < nk; kt++) {
        const int cur = kt & 1;
        wait_async0();          // this wave's staged tile has landed
        __syncthreads();        // everyone's tile is visible
        if (kt + 1 < nk) stage(1 - cur, (kt + 1) * 32);   // overlap next stage

        bf16x16 af[2], bfm[4];
        #pragma unroll
        for (int t = 0; t < 2; t++)
            af[t]  = load_frag(&Al[cur][(wm * 32 + t * 16 + r) * 40], c0, c1);
        #pragma unroll
        for (int t = 0; t < 4; t++)
            bfm[t] = load_frag(&Bl[cur][(wn * 64 + t * 16 + r) * 40], c0, c1);

        #pragma unroll
        for (int i = 0; i < 2; i++)
            #pragma unroll
            for (int j = 0; j < 4; j++)
                acc[i][j] = __builtin_amdgcn_wmma_f32_16x16x32_bf16(
                    false, af[i], false, bfm[j], (short)0, acc[i][j], false, false);
        // buffer `cur` is only overwritten after the *next* iteration's barrier
    }

    // epilogue: bias + store (C layout: element (vgpr v, lane) = row v+8*half, col r)
    #pragma unroll
    for (int i = 0; i < 2; i++) {
        #pragma unroll
        for (int j = 0; j < 4; j++) {
            int baseM = bm * 128 + wm * 32 + i * 16;
            int baseN = bn * 128 + wn * 64 + j * 16;
            #pragma unroll
            for (int v = 0; v < 8; v++) {
                int m = baseM + v + 8 * half;
                int n = baseN + r;
                float val = acc[i][j][v] + bias[n];
                if (mode == 0) {
                    // m = s*B + b  (M == 8192), n = h*64 + d (N == 1024)
                    int s = m >> 3, b = m & 7;
                    int h = n >> 6, d = n & 63;
                    bf16_dst[(((long long)b * H_ + h) * S_ + s) * HD_ + d] = f2bf(val);
                } else {
                    if (m < M) fout[(long long)m * N + n] = val;
                }
            }
        }
    }
}

// ---------------------------------------------------------------------------
// Kernel 3: write dummy last row (raw query[-1]) into Q/K/V at s = S-1
// ---------------------------------------------------------------------------
__global__ void fill_dummy(const float* __restrict__ query,
                           unsigned short* __restrict__ q,
                           unsigned short* __restrict__ k,
                           unsigned short* __restrict__ v) {
    int i = blockIdx.x * blockDim.x + threadIdx.x;   // 0 .. B*E-1
    if (i >= B_ * E_) return;
    int b = i >> 10, e = i & 1023, h = e >> 6, d = e & 63;
    unsigned short x = f2bf(query[((long long)(S_ - 1) * B_ + b) * E_ + e]);
    long long off = (((long long)b * H_ + h) * S_ + (S_ - 1)) * HD_ + d;
    q[off] = x; k[off] = x; v[off] = x;
}

// ---------------------------------------------------------------------------
// Kernel 4: flash-attention per (b,h).  Block = 4 waves, 64 query rows.
// ---------------------------------------------------------------------------
__global__ __launch_bounds__(128)
void attn_kernel(const unsigned short* __restrict__ Q,
                 const unsigned short* __restrict__ Kb,
                 const unsigned short* __restrict__ Vb,
                 unsigned short* __restrict__ Ao) {
    __shared__ __attribute__((aligned(16))) unsigned short Kl[64 * 72];      // [key][d]
    __shared__ __attribute__((aligned(16))) unsigned short Vl[64 * 72];      // [d][key]
    __shared__ __attribute__((aligned(16))) unsigned short Pl[4][16 * 72];   // per wave

    const int tid  = threadIdx.x;
    const int lane = tid & 31;
    const int wave = tid >> 5;
    const int bh   = blockIdx.y;
    const int b    = bh >> 4;
    const int h    = bh & 15;
    const int mblk = blockIdx.x;
    const int r    = lane & 15;
    const int half = lane >> 4;

    const long long base = ((long long)b * H_ + h) * S_ * HD_;
    const int qr0 = mblk * 64 + wave * 16;

    // Q fragments (persist across all key blocks), 2 K-steps over hd=64
    int qrow = qr0 + r; if (qrow >= S_) qrow = S_ - 1;
    bf16x16 qf[2];
    #pragma unroll
    for (int c = 0; c < 2; c++)
        qf[c] = load_frag(&Q[base + (long long)qrow * HD_],
                          c * 32 + half * 8, c * 32 + half * 8 + 16);

    const float scale = 0.125f;  // 1/sqrt(64)
    float mrow[8], lrow[8], corr[8];
    f32x8 zero8 = {0.f,0.f,0.f,0.f,0.f,0.f,0.f,0.f};
    f32x8 oacc[4];
    #pragma unroll
    for (int j = 0; j < 8; j++) { mrow[j] = -1e30f; lrow[j] = 0.f; }
    #pragma unroll
    for (int t = 0; t < 4; t++) oacc[t] = zero8;

    const unsigned klBase = lds_addr_of(&Kl[0]);
    const int nblocks = (S_ + 63) / 64;   // 17
    for (int t = 0; t < nblocks; t++) {
        const int kb0 = t * 64;
        // stage K tile [key][d] with async copies; V tile transposed [d][key] sync
        #pragma unroll
        for (int i = 0; i < 4; i++) {
            int c = tid + i * 128;
            int row = c >> 3, col = (c & 7) * 8;
            int gk = kb0 + row; if (gk >= S_) gk = S_ - 1;
            async_ld_b128(klBase + (unsigned)(row * 72 + col) * 2,
                          &Kb[base + (long long)gk * HD_ + col]);
            u16x8 vv = *(const u16x8*)(&Vb[base + (long long)gk * HD_ + col]);
            #pragma unroll
            for (int e = 0; e < 8; e++) Vl[(col + e) * 72 + row] = vv[e];
        }
        if (t + 1 < nblocks) {
            int gk = kb0 + 64 + (tid >> 1); if (gk >= S_) gk = S_ - 1;
            __builtin_prefetch((const void*)&Vb[base + (long long)gk * HD_]);
        }
        wait_async0();
        __syncthreads();

        // scores: 4 tiles of 16x16, contraction over hd=64 (2 WMMA each)
        f32x8 sc[4];
        #pragma unroll
        for (int tj = 0; tj < 4; tj++) {
            f32x8 a = zero8;
            #pragma unroll
            for (int c = 0; c < 2; c++) {
                bf16x16 kf = load_frag(&Kl[(tj * 16 + r) * 72],
                                       c * 32 + half * 8, c * 32 + half * 8 + 16);
                a = __builtin_amdgcn_wmma_f32_16x16x32_bf16(
                        false, qf[c], false, kf, (short)0, a, false, false);
            }
            sc[tj] = a;
        }

        // scale + OOB key mask
        #pragma unroll
        for (int tj = 0; tj < 4; tj++) {
            bool oob = (kb0 + tj * 16 + r) >= S_;
            #pragma unroll
            for (int v = 0; v < 8; v++)
                sc[tj][v] = oob ? -1e30f : sc[tj][v] * scale;
        }

        // online softmax per row j (row = j + 8*half; stats replicated over 16 lanes)
        #pragma unroll
        for (int j = 0; j < 8; j++) {
            float mx = -1e30f;
            #pragma unroll
            for (int tj = 0; tj < 4; tj++) mx = fmaxf(mx, sc[tj][j]);
            for (int msk = 1; msk < 16; msk <<= 1) mx = fmaxf(mx, __shfl_xor(mx, msk, 32));
            float newm = fmaxf(mrow[j], mx);
            corr[j] = __expf(mrow[j] - newm);
            mrow[j] = newm;
            float rs = 0.f;
            #pragma unroll
            for (int tj = 0; tj < 4; tj++) {
                float p = __expf(sc[tj][j] - newm);
                sc[tj][j] = p;
                rs += p;
            }
            for (int msk = 1; msk < 16; msk <<= 1) rs += __shfl_xor(rs, msk, 32);
            lrow[j] = lrow[j] * corr[j] + rs;
        }

        // rescale running output
        #pragma unroll
        for (int td = 0; td < 4; td++)
            #pragma unroll
            for (int j = 0; j < 8; j++) oacc[td][j] *= corr[j];

        // P (C layout) -> LDS -> A-layout fragments (same-wave DS is in-order)
        #pragma unroll
        for (int tj = 0; tj < 4; tj++)
            #pragma unroll
            for (int j = 0; j < 8; j++)
                Pl[wave][(j + 8 * half) * 72 + tj * 16 + r] = f2bf(sc[tj][j]);

        // O += P @ V   (A = P 16x64, B = V 64x64 via Vl[d][key])
        #pragma unroll
        for (int c = 0; c < 2; c++) {
            bf16x16 pf = load_frag(&Pl[wave][r * 72],
                                   c * 32 + half * 8, c * 32 + half * 8 + 16);
            #pragma unroll
            for (int td = 0; td < 4; td++) {
                bf16x16 vf = load_frag(&Vl[(td * 16 + r) * 72],
                                       c * 32 + half * 8, c * 32 + half * 8 + 16);
                oacc[td] = __builtin_amdgcn_wmma_f32_16x16x32_bf16(
                               false, pf, false, vf, (short)0, oacc[td], false, false);
            }
        }
        __syncthreads();
    }

    // normalize + store bf16 to [S][B][E]
    #pragma unroll
    for (int td = 0; td < 4; td++) {
        #pragma unroll
        for (int j = 0; j < 8; j++) {
            int qr = qr0 + j + 8 * half;
            if (qr >= S_) continue;
            int d = td * 16 + r;
            float val = oacc[td][j] / lrow[j];
            Ao[((long long)qr * B_ + b) * E_ + h * HD_ + d] = f2bf(val);
        }
    }
}

// ---------------------------------------------------------------------------
// Host launcher
// ---------------------------------------------------------------------------
extern "C" void kernel_launch(void* const* d_in, const int* in_sizes, int n_in,
                              void* d_out, int out_size, void* d_ws, size_t ws_size,
                              hipStream_t stream) {
    const float* query = (const float*)d_in[0];
    const float* key   = (const float*)d_in[1];
    const float* value = (const float*)d_in[2];
    const float* w_in  = (const float*)d_in[3];   // (3E, E)
    const float* b_in  = (const float*)d_in[4];   // (3E,)
    const float* w_out = (const float*)d_in[5];   // (E, E)
    const float* b_out = (const float*)d_in[6];   // (E,)
    float* out = (float*)d_out;

    char* ws = (char*)d_ws;
    auto alloc = [&](size_t bytes) -> char* {
        char* p = ws;
        ws += (bytes + 255) & ~(size_t)255;
        return p;
    };

    const long long MR = 8192;            // rows used for projections (s < 1024)
    unsigned short* Xq  = (unsigned short*)alloc(MR * E_ * 2);
    unsigned short* Xk  = (unsigned short*)alloc(MR * E_ * 2);
    unsigned short* Xv  = (unsigned short*)alloc(MR * E_ * 2);
    unsigned short* Wi  = (unsigned short*)alloc(3LL * E_ * E_ * 2);
    unsigned short* Wo  = (unsigned short*)alloc((long long)E_ * E_ * 2);
    const long long QSZ = (long long)B_ * H_ * S_ * HD_;
    unsigned short* Qb  = (unsigned short*)alloc(QSZ * 2);
    unsigned short* Kbf = (unsigned short*)alloc(QSZ * 2);
    unsigned short* Vbf = (unsigned short*)alloc(QSZ * 2);
    unsigned short* Ao  = (unsigned short*)alloc((long long)S_ * B_ * E_ * 2);

    // 1) convert activations + weights to bf16
    convert_f32_bf16<<<4096, 256, 0, stream>>>(query, Xq, MR * E_);
    convert_f32_bf16<<<4096, 256, 0, stream>>>(key,   Xk, MR * E_);
    convert_f32_bf16<<<4096, 256, 0, stream>>>(value, Xv, MR * E_);
    convert_f32_bf16<<<4096, 256, 0, stream>>>(w_in,  Wi, 3LL * E_ * E_);
    convert_f32_bf16<<<4096, 256, 0, stream>>>(w_out, Wo, (long long)E_ * E_);

    // 2) Q/K/V projections: M=8192, N=1024, K=1024 each
    dim3 gp(8192 / 128, 1024 / 128);
    gemm_bf16<<<gp, 256, 0, stream>>>(Xq, Wi,                 b_in,
                                      8192, 1024, 1024, Qb,  nullptr, 0);
    gemm_bf16<<<gp, 256, 0, stream>>>(Xk, Wi + 1024LL * 1024, b_in + 1024,
                                      8192, 1024, 1024, Kbf, nullptr, 0);
    gemm_bf16<<<gp, 256, 0, stream>>>(Xv, Wi + 2048LL * 1024, b_in + 2048,
                                      8192, 1024, 1024, Vbf, nullptr, 0);

    // 3) dummy row s = S-1 (raw query[-1] into all three)
    fill_dummy<<<(B_ * E_ + 255) / 256, 256, 0, stream>>>(query, Qb, Kbf, Vbf);

    // 4) attention: grid (17 query tiles, B*H heads)
    dim3 ga((S_ + 63) / 64, B_ * H_);
    attn_kernel<<<ga, 128, 0, stream>>>(Qb, Kbf, Vbf, Ao);

    // 5) output projection: M = S*B = 8200, N = K = 1024
    dim3 go((S_ * B_ + 127) / 128, 1024 / 128);
    gemm_bf16<<<go, 256, 0, stream>>>(Ao, Wo, b_out,
                                      S_ * B_, 1024, 1024, nullptr, out, 1);
}
